// LinearAttention_85667417686560
// MI455X (gfx1250) — matrix operand order repaired and checked by reading, not compiled, not run on hardware
//
#include <hip/hip_runtime.h>

typedef _Float16 half_t;
typedef _Float16 v16h __attribute__((ext_vector_type(16)));
typedef _Float16 v8h  __attribute__((ext_vector_type(8)));
typedef float    v8f  __attribute__((ext_vector_type(8)));
typedef float    v4f  __attribute__((ext_vector_type(4)));
typedef unsigned int u32x4 __attribute__((ext_vector_type(4)));
typedef int          i32x4 __attribute__((ext_vector_type(4)));
typedef int          i32x8 __attribute__((ext_vector_type(8)));

#define NB 4
#define NT 8192
#define NE 1024
#define NH 16
#define ND 64
#define NM (NB * NT)           // 32768 rows
#define SCALE_F 0.125f         // 1/sqrt(64)
#define EPS_FM 1e-6f
#define EPS_Z 1e-8f
#define LPITCH 72              // LDS tile pitch in halves (144B rows: conflict-free)

static __device__ __forceinline__ v8f wmma_f16(v16h a, v16h b, v8f c) {
  return __builtin_amdgcn_wmma_f32_16x16x32_f16(false, a, false, b, (short)0, c,
                                                false, false);
}

// ---- Fragment loaders (global) ---------------------------------------------
// CDNA5 16-bit A fragment (16x32 MxK), row-major A, lda in elements.
static __device__ __forceinline__ v16h load_a_f16(const half_t* A, size_t lda,
                                                  int m0, int k0, int lane) {
  int m = lane & 15, ph = lane >> 4;
  const half_t* p = A + (size_t)(m0 + m) * lda + (size_t)k0 + ph * 8;
  union { v16h v; v8h h[2]; } u;
  u.h[0] = *(const v8h*)(p);
  u.h[1] = *(const v8h*)(p + 16);
  return u.v;
}

static __device__ __forceinline__ v16h load_a_f32cvt(const float* A, size_t lda,
                                                     int m0, int k0, int lane) {
  int m = lane & 15, ph = lane >> 4;
  const float* p = A + (size_t)(m0 + m) * lda + (size_t)k0 + ph * 8;
  const v4f* q = (const v4f*)p;
  v4f x0 = q[0], x1 = q[1], x2 = q[4], x3 = q[5];
  v16h f;
#pragma unroll
  for (int i = 0; i < 4; ++i) {
    f[i]      = (_Float16)x0[i];
    f[4 + i]  = (_Float16)x1[i];
    f[8 + i]  = (_Float16)x2[i];
    f[12 + i] = (_Float16)x3[i];
  }
  return f;
}

// B fragment (32x16 KxN) from Bt stored row-major N x K  -> computes A * Bt^T
static __device__ __forceinline__ v16h load_bt_f16(const half_t* Bt, size_t ldb,
                                                   int n0, int k0, int lane) {
  int n = lane & 15, ph = lane >> 4;
  const half_t* p = Bt + (size_t)(n0 + n) * ldb + (size_t)k0 + ph * 8;
  union { v16h v; v8h h[2]; } u;
  u.h[0] = *(const v8h*)(p);
  u.h[1] = *(const v8h*)(p + 16);
  return u.v;
}

// ---- Fragment loaders (LDS tile, pitch LPITCH halves) -----------------------
static __device__ __forceinline__ v16h lds_frag_a(const half_t* base, int m0,
                                                  int k0, int lane) {
  int m = lane & 15, ph = lane >> 4;
  const half_t* p = base + (m0 + m) * LPITCH + k0 + ph * 8;
  union { v16h v; v8h h[2]; } u;
  u.h[0] = *(const v8h*)(p);
  u.h[1] = *(const v8h*)(p + 16);
  return u.v;
}
static __device__ __forceinline__ v16h lds_frag_b(const half_t* base, int n0,
                                                  int k0, int lane) {
  int n = lane & 15, ph = lane >> 4;
  const half_t* p = base + (n0 + n) * LPITCH + k0 + ph * 8;
  union { v16h v; v8h h[2]; } u;
  u.h[0] = *(const v8h*)(p);
  u.h[1] = *(const v8h*)(p + 16);
  return u.v;
}

// ---- CDNA5 async / TDM primitives ------------------------------------------
static __device__ __forceinline__ void async_b128(void* lds_dst, const void* g) {
  asm volatile("global_load_async_to_lds_b128 %0, %1, off" ::
                   "v"((unsigned int)(uintptr_t)lds_dst),
                   "v"((unsigned long long)(uintptr_t)g)
               : "memory");
}
static __device__ __forceinline__ void wait_async0() {
  asm volatile("s_wait_asynccnt 0x0" ::: "memory");
}

// TDM: 2D tile load (tile_d0 x tile_d1 elements of 2B) into LDS with padding
// of 4 DWORDs after every 32 DWORDs (-> LPITCH=72 halves for tile_d0=64).
static __device__ __forceinline__ void tdm_load_2d(void* lds_dst, const void* g,
                                                   unsigned tensor_d0,
                                                   unsigned tensor_d1,
                                                   unsigned stride0,
                                                   unsigned tile_d0,
                                                   unsigned tile_d1) {
  unsigned long long ga = (unsigned long long)(uintptr_t)g;
  u32x4 g0;
  g0[0] = 1u;                                   // count=1, user descriptor
  g0[1] = (unsigned int)(uintptr_t)lds_dst;     // lds_addr
  g0[2] = (unsigned int)ga;                     // global_addr[31:0]
  g0[3] = (unsigned int)((ga >> 32) & 0x1ffffffu) | (2u << 30);  // addr hi, type=2
  i32x8 g1;
  g1[0] = (int)((1u << 16)      // data_size = 2 bytes
              | (1u << 20)      // pad_enable
              | (4u << 22)      // pad_interval = 32 dwords
              | (3u << 25));    // pad_amount   = 4 dwords (8 halves)
  g1[1] = (int)((tensor_d0 & 0xffffu) << 16);
  g1[2] = (int)(((tensor_d0 >> 16) & 0xffffu) | ((tensor_d1 & 0xffffu) << 16));
  g1[3] = (int)(((tensor_d1 >> 16) & 0xffffu) | ((tile_d0 & 0xffffu) << 16));
  g1[4] = (int)(tile_d1 & 0xffffu);             // tile_dim1 (tile_dim2 = 0)
  g1[5] = (int)stride0;                         // tensor_dim0_stride lo
  g1[6] = 0;
  g1[7] = 0;
  i32x4 z4 = {0, 0, 0, 0};
#if __clang_major__ >= 23
  i32x8 z8 = {0, 0, 0, 0, 0, 0, 0, 0};
  __builtin_amdgcn_tensor_load_to_lds(g0, g1, z4, z4, z8, 0);
#else
  __builtin_amdgcn_tensor_load_to_lds(g0, g1, z4, z4, 0);
#endif
}

// ---- Kernel 0: fp32 -> f16 weight conversion --------------------------------
__global__ __launch_bounds__(256) void cvt_kernel(const float* __restrict__ in,
                                                  half_t* __restrict__ out,
                                                  int n) {
  int i = (blockIdx.x * 256 + threadIdx.x) * 4;
  if (i + 3 < n) {
    v4f x = *(const v4f*)(in + i);
#pragma unroll
    for (int j = 0; j < 4; ++j) out[i + j] = (half_t)x[j];
  }
}

// ---- Kernel 1: projection GEMM  Out = epilogue(X @ W^T + b) -----------------
// MODE 0: Q -> relu(y*SCALE)+eps, store (B,H,T,D)
// MODE 1: K -> relu(y*SCALE)+eps, store (B,H,D,T)  (transposed)
// MODE 2: V -> y,                 store (B,H,D,T)  (transposed)
template <int MODE>
__global__ __launch_bounds__(256) void proj_kernel(
    const float* __restrict__ X, const half_t* __restrict__ W,
    const float* __restrict__ bias, half_t* __restrict__ Out) {
  const int lane = threadIdx.x & 31;
  const int wave = threadIdx.x >> 5;
  const int m0 = blockIdx.x * 256 + (wave >> 1) * 64;
  const int n0 = blockIdx.y * 128 + (wave & 1) * 64;

  v8f acc[4][4] = {};
  for (int k0 = 0; k0 < NE; k0 += 32) {
    v16h a[4], b[4];
#pragma unroll
    for (int i = 0; i < 4; ++i) a[i] = load_a_f32cvt(X, NE, m0 + 16 * i, k0, lane);
#pragma unroll
    for (int j = 0; j < 4; ++j) b[j] = load_bt_f16(W, NE, n0 + 16 * j, k0, lane);
    __builtin_prefetch(X + (size_t)(m0 + (lane & 15)) * NE + k0 + 64, 0, 1);
#pragma unroll
    for (int i = 0; i < 4; ++i)
#pragma unroll
      for (int j = 0; j < 4; ++j) acc[i][j] = wmma_f16(a[i], b[j], acc[i][j]);
  }

  const int ln = lane & 15, lh = lane >> 4;
#pragma unroll
  for (int j = 0; j < 4; ++j) {
    const int n = n0 + 16 * j + ln;
    const float bv = bias[n];
    const int h = n >> 6, d = n & 63;
#pragma unroll
    for (int i = 0; i < 4; ++i) {
#pragma unroll
      for (int r = 0; r < 8; ++r) {
        const int m = m0 + 16 * i + r + 8 * lh;
        float y = acc[i][j][r] + bv;
        if (MODE < 2) {
          y *= SCALE_F;
          y = y > 0.0f ? y : 0.0f;
          y += EPS_FM;
        }
        const int bb = m >> 13;
        const int t = m & (NT - 1);
        size_t o;
        if (MODE == 0)
          o = ((size_t)(bb * NH + h) * NT + t) * ND + d;     // (B,H,T,D)
        else
          o = ((size_t)(bb * NH + h) * ND + d) * NT + t;     // (B,H,D,T)
        Out[o] = (half_t)y;
      }
    }
  }
}

// ---- Kernel 2: per-(b,h)  KV = K^T V  + K_sum -------------------------------
// TDM double-buffered, 128-wide T-chunks as 2 x (64x64) padded sub-tiles.
__global__ __launch_bounds__(256) void kv_kernel(const half_t* __restrict__ Kt,
                                                 const half_t* __restrict__ Vt,
                                                 half_t* __restrict__ KVt,
                                                 float* __restrict__ Ksum) {
  const int bh = blockIdx.x;  // 0..63
  const half_t* Kh = Kt + (size_t)bh * ND * NT;
  const half_t* Vh = Vt + (size_t)bh * ND * NT;
  const int lane = threadIdx.x & 31;
  const int wave = threadIdx.x >> 5;
  const int ti = wave & 3;          // d-subtile
  const int tj = (wave >> 2) * 2;   // e-subtile pair

  __shared__ half_t tK[2][2][ND * LPITCH];   // [buf][sub]
  __shared__ half_t tV[2][2][ND * LPITCH];
  __shared__ float red[256];

  const bool leader = (__builtin_amdgcn_readfirstlane(wave) == 0);

  if (leader) {
#pragma unroll
    for (int s = 0; s < 2; ++s) {
      tdm_load_2d(&tK[0][s][0], Kh + s * 64, NT, ND, NT, 64, 64);
      tdm_load_2d(&tV[0][s][0], Vh + s * 64, NT, ND, NT, 64, 64);
    }
  }

  v8f acc0 = {}, acc1 = {};
  float ksp = 0.0f;                     // K_sum partial (row krow, slice kq)
  const int krow = threadIdx.x >> 2;    // 0..63
  const int kq = threadIdx.x & 3;

  int cur = 0;
  const int NCH = NT / 128;             // 64 chunks
  for (int c = 0; c < NCH; ++c) {
    if (leader) {
      if (c + 1 < NCH) {
#pragma unroll
        for (int s = 0; s < 2; ++s) {
          const size_t off = (size_t)(c + 1) * 128 + s * 64;
          tdm_load_2d(&tK[cur ^ 1][s][0], Kh + off, NT, ND, NT, 64, 64);
          tdm_load_2d(&tV[cur ^ 1][s][0], Vh + off, NT, ND, NT, 64, 64);
        }
        __builtin_amdgcn_s_wait_tensorcnt((short)4);  // current buffer done
      } else {
        __builtin_amdgcn_s_wait_tensorcnt((short)0);
      }
    }
    __syncthreads();
#pragma unroll
    for (int s = 0; s < 2; ++s) {
      const half_t* bK = &tK[cur][s][0];
      const half_t* bV = &tV[cur][s][0];
#pragma unroll
      for (int kk = 0; kk < 64; kk += 32) {
        v16h a = lds_frag_a(bK, ti * 16, kk, lane);
        v16h b0 = lds_frag_b(bV, tj * 16, kk, lane);
        v16h b1 = lds_frag_b(bV, (tj + 1) * 16, kk, lane);
        acc0 = wmma_f16(a, b0, acc0);
        acc1 = wmma_f16(a, b1, acc1);
      }
      // K_sum partial from the LDS tile (co-executes with WMMA)
      const half_t* pr = &tK[cur][s][krow * LPITCH + kq * 16];
#pragma unroll
      for (int e = 0; e < 16; ++e) ksp += (float)pr[e];
    }
    __syncthreads();
    cur ^= 1;
  }

  // store KV transposed (e-major, d contiguous) for stage-3 B fragments
  const int ln = lane & 15, lh = lane >> 4;
  half_t* O = KVt + (size_t)bh * ND * ND;
#pragma unroll
  for (int r = 0; r < 8; ++r) {
    const int d = ti * 16 + r + 8 * lh;
    const int e = tj * 16 + ln;
    O[(size_t)e * ND + d] = (half_t)acc0[r];
    O[(size_t)(e + 16) * ND + d] = (half_t)acc1[r];
  }

  red[threadIdx.x] = ksp;
  __syncthreads();
  if (kq == 0)
    Ksum[bh * ND + krow] = red[threadIdx.x] + red[threadIdx.x + 1] +
                           red[threadIdx.x + 2] + red[threadIdx.x + 3];
}

// ---- Kernel 3: Att = (Q @ KV) / (Q . K_sum + eps) ---------------------------
__global__ __launch_bounds__(256) void qkv_kernel(const half_t* __restrict__ Qf,
                                                  const half_t* __restrict__ KVt,
                                                  const float* __restrict__ Ksum,
                                                  half_t* __restrict__ Att) {
  const int bh = blockIdx.y;
  const int b = bh >> 4, h = bh & 15;
  const int t0 = blockIdx.x * 128;
  const half_t* Qh = Qf + (size_t)bh * NT * ND;
  const half_t* KV = KVt + (size_t)bh * ND * ND;

  __shared__ half_t kvs[ND * LPITCH];   // staged KV^T tile, padded pitch
  __shared__ float ks[ND];
  __shared__ float zinv[128];

  // async-copy the 8KB KV tile into LDS (2 x 16B per thread)
#pragma unroll
  for (int it = 0; it < 2; ++it) {
    const int idx = threadIdx.x + 256 * it;   // 0..511 -> row 0..63, seg 0..7
    const int row = idx >> 3, seg = idx & 7;
    async_b128(&kvs[row * LPITCH + seg * 8], KV + row * ND + seg * 8);
  }
  if (threadIdx.x < ND) ks[threadIdx.x] = Ksum[bh * ND + threadIdx.x];
  wait_async0();
  __syncthreads();

  if (threadIdx.x < 128) {
    const half_t* qr = Qh + (size_t)(t0 + threadIdx.x) * ND;
    float sdot = 0.0f;
    for (int d = 0; d < ND; d += 8) {
      v8h vv = *(const v8h*)(qr + d);
#pragma unroll
      for (int e = 0; e < 8; ++e) sdot += (float)vv[e] * ks[d + e];
    }
    zinv[threadIdx.x] = 1.0f / (sdot + EPS_Z);
  }
  __syncthreads();

  const int lane = threadIdx.x & 31;
  const int wave = threadIdx.x >> 5;   // 8 waves -> 8 x 16 rows
  const int m0 = wave * 16;
  v8f acc[4] = {};
  for (int k0 = 0; k0 < ND; k0 += 32) {
    v16h a = load_a_f16(Qh, ND, t0 + m0, k0, lane);
#pragma unroll
    for (int j = 0; j < 4; ++j) {
      v16h bf = lds_frag_b(kvs, j * 16, k0, lane);
      acc[j] = wmma_f16(a, bf, acc[j]);
    }
  }
  const int ln = lane & 15, lh = lane >> 4;
#pragma unroll
  for (int j = 0; j < 4; ++j) {
    const int e = j * 16 + ln;
#pragma unroll
    for (int r = 0; r < 8; ++r) {
      const int ml = m0 + r + 8 * lh;
      const float y = acc[j][r] * zinv[ml];
      Att[((size_t)(b * NT + t0 + ml)) * NE + h * ND + e] = (half_t)y;
    }
  }
}

// ---- Kernel 4: final projection  Out = Att @ Wo^T + bo  (fp32 out) ----------
__global__ __launch_bounds__(256) void oproj_kernel(
    const half_t* __restrict__ Att, const half_t* __restrict__ W,
    const float* __restrict__ bias, float* __restrict__ Out) {
  const int lane = threadIdx.x & 31;
  const int wave = threadIdx.x >> 5;
  const int m0 = blockIdx.x * 256 + (wave >> 1) * 64;
  const int n0 = blockIdx.y * 128 + (wave & 1) * 64;

  v8f acc[4][4] = {};
  for (int k0 = 0; k0 < NE; k0 += 32) {
    v16h a[4], b[4];
#pragma unroll
    for (int i = 0; i < 4; ++i) a[i] = load_a_f16(Att, NE, m0 + 16 * i, k0, lane);
#pragma unroll
    for (int j = 0; j < 4; ++j) b[j] = load_bt_f16(W, NE, n0 + 16 * j, k0, lane);
    __builtin_prefetch(Att + (size_t)(m0 + (lane & 15)) * NE + k0 + 64, 0, 1);
#pragma unroll
    for (int i = 0; i < 4; ++i)
#pragma unroll
      for (int j = 0; j < 4; ++j) acc[i][j] = wmma_f16(a[i], b[j], acc[i][j]);
  }
  const int ln = lane & 15, lh = lane >> 4;
#pragma unroll
  for (int j = 0; j < 4; ++j) {
    const int n = n0 + 16 * j + ln;
    const float bv = bias[n];
#pragma unroll
    for (int i = 0; i < 4; ++i) {
#pragma unroll
      for (int r = 0; r < 8; ++r) {
        const int m = m0 + 16 * i + r + 8 * lh;
        __builtin_nontemporal_store(acc[i][j][r] + bv, Out + (size_t)m * NE + n);
      }
    }
  }
}

// ---- Host launch ------------------------------------------------------------
extern "C" void kernel_launch(void* const* d_in, const int* in_sizes, int n_in,
                              void* d_out, int out_size, void* d_ws,
                              size_t ws_size, hipStream_t stream) {
  const float* q  = (const float*)d_in[0];
  const float* k  = (const float*)d_in[1];
  const float* v  = (const float*)d_in[2];
  const float* Wq = (const float*)d_in[3];
  const float* bq = (const float*)d_in[4];
  const float* Wk = (const float*)d_in[5];
  const float* bk = (const float*)d_in[6];
  const float* Wv = (const float*)d_in[7];
  const float* bv = (const float*)d_in[8];
  const float* Wo = (const float*)d_in[9];
  const float* bo = (const float*)d_in[10];

  const size_t EW = (size_t)NE * NE;
  const size_t EA = (size_t)NM * NE;
  half_t* Wq16 = (half_t*)d_ws;
  half_t* Wk16 = Wq16 + EW;
  half_t* Wv16 = Wk16 + EW;
  half_t* Wo16 = Wv16 + EW;
  half_t* Qf   = Wo16 + EW;                 // (B,H,T,D)
  half_t* Kt   = Qf + EA;                   // (B,H,D,T)
  half_t* Vt   = Kt + EA;                   // (B,H,D,T)
  half_t* Att  = Vt + EA;                   // (B,T,E)
  half_t* KVt  = Att + EA;                  // (B*H,64,64) e-major
  float*  Ksum = (float*)(KVt + (size_t)NB * NH * ND * ND);

  cvt_kernel<<<1024, 256, 0, stream>>>(Wq, Wq16, (int)EW);
  cvt_kernel<<<1024, 256, 0, stream>>>(Wk, Wk16, (int)EW);
  cvt_kernel<<<1024, 256, 0, stream>>>(Wv, Wv16, (int)EW);
  cvt_kernel<<<1024, 256, 0, stream>>>(Wo, Wo16, (int)EW);

  dim3 gproj(NM / 256, NE / 128);
  proj_kernel<0><<<gproj, 256, 0, stream>>>(q, Wq16, bq, Qf);
  proj_kernel<1><<<gproj, 256, 0, stream>>>(k, Wk16, bk, Kt);
  proj_kernel<2><<<gproj, 256, 0, stream>>>(v, Wv16, bv, Vt);

  kv_kernel<<<NB * NH, 256, 0, stream>>>(Kt, Vt, KVt, Ksum);

  dim3 gqkv(NT / 128, NB * NH);
  qkv_kernel<<<gqkv, 256, 0, stream>>>(Qf, KVt, Ksum, Att);

  oproj_kernel<<<gproj, 256, 0, stream>>>(Att, Wo16, bo, (float*)d_out);
}
